// RubyComplexityGNN_22565758173966
// MI455X (gfx1250) — matrix operand order repaired and checked by reading, not compiled, not run on hardware
//
#include <hip/hip_runtime.h>
#include <hip/hip_fp16.h>

#define DIM 128

typedef __attribute__((ext_vector_type(16))) _Float16 v16h;
typedef __attribute__((ext_vector_type(8)))  _Float16 v8h;
typedef __attribute__((ext_vector_type(8)))  float    v8f;
typedef __attribute__((ext_vector_type(4)))  int      v4i;

union AFrag { v16h v; v8h h[2]; };
union BFrag { v16h v; v8h h[2]; };
union CFrag { v8f  v; float   e[8]; };

#if __has_builtin(__builtin_amdgcn_global_load_async_to_lds_b128)
#define HAVE_ASYNC_LDS 1
typedef __attribute__((address_space(1))) v4i gv4i;   // global 16B chunk
typedef __attribute__((address_space(3))) v4i lv4i;   // LDS 16B chunk
#endif

// ---------------- degree / norm ----------------
__global__ void k_init_deg(float* deg, int n) {
    int i = blockIdx.x * blockDim.x + threadIdx.x;
    if (i < n) deg[i] = 1.0f;                    // self-loop contributes 1
}

__global__ void k_deg_accum(const int* __restrict__ dsts, float* __restrict__ deg, int e) {
    int i = blockIdx.x * blockDim.x + threadIdx.x;
    if (i < e) atomicAdd(&deg[dsts[i]], 1.0f);
}

__global__ void k_rsqrt_inplace(float* deg, int n) {
    int i = blockIdx.x * blockDim.x + threadIdx.x;
    if (i < n) deg[i] = rsqrtf(fmaxf(deg[i], 1.0f));
}

// ---------------- f32 -> f16 (optionally fused ReLU) ----------------
template <bool RELU>
__global__ void k_f32_to_f16(const float* __restrict__ in, _Float16* __restrict__ out, int n) {
    int i = blockIdx.x * blockDim.x + threadIdx.x;
    if (i < n) {
        float v = in[i];
        if (RELU) v = fmaxf(v, 0.0f);
        out[i] = (_Float16)v;
    }
}

// ---------------- pack W[128x128] f32 into WMMA-B fragment order (f16) ----------------
// Bpk[((kk*2+half)*128 + col)*16 + j] = W[(kk*32 + half*16 + j)*128 + col]
// => each lane's 16-element B fragment is 32 contiguous bytes.
__global__ void k_pack_w(const float* __restrict__ W, _Float16* __restrict__ Bpk) {
    int o = blockIdx.x * blockDim.x + threadIdx.x;   // 16384
    int j  = o & 15;
    int n  = (o >> 4) & (DIM - 1);
    int hh = (o >> 11) & 1;
    int kk = o >> 12;
    int k  = kk * 32 + hh * 16 + j;
    Bpk[o] = (_Float16)W[(size_t)k * DIM + n];
}

// ---------------- GEMM + fused self-loop/bias epilogue ----------------
// One wave computes a 16-row x 64-col tile: 4 accumulators, 4 K-chunks ->
// 16 v_wmma_f32_16x16x32_f16 per wave. All 4 B fragments of a K-chunk are
// loaded into distinct register sets before the WMMAs so ds latency overlaps.
// Packed B (32KB) staged once per block in LDS via async-to-LDS.
__global__ void __launch_bounds__(256)
k_gemm_strip(const _Float16* __restrict__ A, const _Float16* __restrict__ Bpk,
             const float* __restrict__ dinv, const float* __restrict__ bias,
             float* __restrict__ T, float* __restrict__ H, int nStrips) {
    __shared__ __align__(16) _Float16 Bs[DIM * DIM];     // 32 KB of 320 KB LDS

    const int tid = threadIdx.x;
    // ---- cooperative stage of packed B: 256 threads x 8 x 16B = 32KB ----
#ifdef HAVE_ASYNC_LDS
    #pragma unroll
    for (int i = 0; i < 8; ++i) {
        const int idx = (tid + i * 256) * 8;             // in halves (16B chunks)
        __builtin_amdgcn_global_load_async_to_lds_b128(
            (gv4i*)(Bpk + idx), (lv4i*)(Bs + idx), 0, 0);
    }
#if __has_builtin(__builtin_amdgcn_s_wait_asynccnt)
    __builtin_amdgcn_s_wait_asynccnt(0);
#else
    asm volatile("s_wait_asynccnt 0" ::: "memory");
#endif
    __syncthreads();
#else
    #pragma unroll
    for (int i = 0; i < 8; ++i) {
        const int idx = (tid + i * 256) * 8;
        *(v8h*)(Bs + idx) = *(const v8h*)(Bpk + idx);
    }
    __syncthreads();
#endif

    const int lane  = tid & 31;
    const int wave  = tid >> 5;
    const int work  = blockIdx.x * 8 + wave;             // wave work item
    const int strip = work >> 1;                         // 16-row strip
    const int nh    = work & 1;                          // column half: [nh*64, nh*64+64)
    if (strip >= nStrips) return;                        // wave-uniform: EXEC all-1 for WMMA
    const int half = lane >> 4;                          // lane group 0-15 / 16-31
    const int mr   = lane & 15;
    const int m    = strip * 16 + mr;                    // A row for this lane

    CFrag acc[4];
    #pragma unroll
    for (int n = 0; n < 4; ++n)
        #pragma unroll
        for (int r = 0; r < 8; ++r) acc[n].e[r] = 0.0f;

    #pragma unroll
    for (int kk = 0; kk < 4; ++kk) {
        // A fragment: lanes 0-15 hold K=kc+[0..15]; lanes 16-31 K=kc+[16..31]
        AFrag a;
        a.h[0] = *(const v8h*)(A + (size_t)m * DIM + kk * 32 + half * 16);
        a.h[1] = *(const v8h*)(A + (size_t)m * DIM + kk * 32 + half * 16 + 8);

        // Load ALL 4 B fragments into distinct register sets (one ds clause),
        // then issue the 4 WMMAs -> partial dscnt waits overlap load latency.
        BFrag b[4];
        const _Float16* bbase = Bs + ((size_t)(kk * 2 + half) * DIM + nh * 64 + mr) * 16;
        #pragma unroll
        for (int n = 0; n < 4; ++n) {
            const _Float16* bp = bbase + (size_t)n * 16 * 16;
            b[n].h[0] = *(const v8h*)(bp);
            b[n].h[1] = *(const v8h*)(bp + 8);
        }
        #pragma unroll
        for (int n = 0; n < 4; ++n)
            acc[n].v = __builtin_amdgcn_wmma_f32_16x16x32_f16(false, a.v, false, b[n].v,
                                                              (short)0, acc[n].v, false, false);
    }

    // ---- epilogue: T = A@W ; H = dinv[row]^2 * T + bias[col] ----
    float dv[8];
    #pragma unroll
    for (int r = 0; r < 8; ++r) {
        const float d = dinv[strip * 16 + half * 8 + r];
        dv[r] = d * d;
    }
    #pragma unroll
    for (int n = 0; n < 4; ++n) {
        const int col = nh * 64 + n * 16 + mr;
        const float bc = bias[col];
        #pragma unroll
        for (int r = 0; r < 8; ++r) {
            const int row = strip * 16 + half * 8 + r;
            const size_t o = (size_t)row * DIM + col;
            const float v = acc[n].e[r];
            T[o] = v;
            H[o] = dv[r] * v + bc;
        }
    }
}

// ---------------- edge scatter: one wave32 per edge, float4 per lane ----------------
__global__ void k_edge_scatter(const float* __restrict__ t,
                               const int* __restrict__ srcs, const int* __restrict__ dsts,
                               const float* __restrict__ dinv, float* __restrict__ h, int e) {
    const int gw   = (blockIdx.x * blockDim.x + threadIdx.x) >> 5;   // edge id
    const int lane = threadIdx.x & 31;
    if (gw >= e) return;
    if (lane == 0 && gw + 4096 < e) {            // keep edge stream hot (global_prefetch_b8)
        __builtin_prefetch(&srcs[gw + 4096], 0, 0);
        __builtin_prefetch(&dsts[gw + 4096], 0, 0);
    }
    const int s = srcs[gw], d = dsts[gw];
    const float w = dinv[s] * dinv[d];
    const float4 v = *(const float4*)(t + (size_t)s * DIM + lane * 4);
    float* o = h + (size_t)d * DIM + lane * 4;
    atomicAdd(o + 0, w * v.x);
    atomicAdd(o + 1, w * v.y);
    atomicAdd(o + 2, w * v.z);
    atomicAdd(o + 3, w * v.w);
}

// ---------------- pooling ----------------
__global__ void k_zero(float* p, int n) {
    int i = blockIdx.x * blockDim.x + threadIdx.x;
    if (i < n) p[i] = 0.0f;
}

__global__ void k_pool_accum(const float* __restrict__ h, const int* __restrict__ batch,
                             float* __restrict__ pool, int nNodes) {
    int gid  = blockIdx.x * blockDim.x + threadIdx.x;
    int node = gid >> 5;
    int c4   = (gid & 31) * 4;
    if (node >= nNodes) return;
    int b = batch[node];
    const float4 v = *(const float4*)(h + (size_t)node * DIM + c4);
    float* o = pool + (size_t)b * DIM + c4;
    atomicAdd(o + 0, v.x);
    atomicAdd(o + 1, v.y);
    atomicAdd(o + 2, v.z);
    atomicAdd(o + 3, v.w);
}

__global__ void k_cnt_accum(const int* __restrict__ batch, float* __restrict__ cnt, int nNodes) {
    int i = blockIdx.x * blockDim.x + threadIdx.x;
    if (i < nNodes) atomicAdd(&cnt[batch[i]], 1.0f);
}

// ---------------- final projection: out[b] = (pool[b]/cnt[b]) . Wp + bp ----------------
__global__ void k_final(const float* __restrict__ pool, const float* __restrict__ cnt,
                        const float* __restrict__ Wp, const float* __restrict__ bp,
                        float* __restrict__ out) {
    __shared__ float red[DIM];
    int b = blockIdx.x, t = threadIdx.x;
    float c = fmaxf(cnt[b], 1.0f);
    red[t] = (pool[(size_t)b * DIM + t] / c) * Wp[t];
    __syncthreads();
    for (int s = DIM / 2; s > 0; s >>= 1) {
        if (t < s) red[t] += red[t + s];
        __syncthreads();
    }
    if (t == 0) out[b] = red[0] + bp[0];
}

static inline int cdiv(int a, int b) { return (a + b - 1) / b; }

extern "C" void kernel_launch(void* const* d_in, const int* in_sizes, int n_in,
                              void* d_out, int out_size, void* d_ws, size_t ws_size,
                              hipStream_t stream) {
    const float* x    = (const float*)d_in[0];
    const int*   ei   = (const int*)d_in[1];
    const int*   batch= (const int*)d_in[2];
    const float* W1   = (const float*)d_in[3];
    const float* b1   = (const float*)d_in[4];
    const float* W2   = (const float*)d_in[5];
    const float* b2   = (const float*)d_in[6];
    const float* W3   = (const float*)d_in[7];
    const float* b3   = (const float*)d_in[8];
    const float* Wp   = (const float*)d_in[9];
    const float* bp   = (const float*)d_in[10];
    float* out = (float*)d_out;

    const int nNodes = in_sizes[0] / DIM;     // 10000
    const int nEdges = in_sizes[1] / 2;       // 640000
    const int nGraph = out_size;              // 64
    const int* srcs = ei;
    const int* dsts = ei + nEdges;

    // workspace layout (256B aligned chunks)
    char* ws = (char*)d_ws;
    size_t off = 0;
    auto alloc = [&](size_t bytes) -> void* {
        void* p = ws + off;
        off += (bytes + 255) & ~(size_t)255;
        return p;
    };
    float*    dinv = (float*)   alloc((size_t)nNodes * 4);
    _Float16* hf16 = (_Float16*)alloc((size_t)nNodes * DIM * 2);
    _Float16* Bpk  = (_Float16*)alloc((size_t)DIM * DIM * 2);
    float*    tbuf = (float*)   alloc((size_t)nNodes * DIM * 4);
    float*    hA   = (float*)   alloc((size_t)nNodes * DIM * 4);
    float*    hB   = (float*)   alloc((size_t)nNodes * DIM * 4);
    float*    pool = (float*)   alloc((size_t)nGraph * DIM * 4);
    float*    cnt  = (float*)   alloc((size_t)nGraph * 4);

    const int totalElems = nNodes * DIM;
    const int nStrips    = cdiv(nNodes, 16);               // 625
    const int gemmBlocks = cdiv(nStrips * 2, 8);           // 2 wave-works per strip, 8 waves/block

    // --- degree / normalization ---
    k_init_deg<<<cdiv(nNodes, 256), 256, 0, stream>>>(dinv, nNodes);
    k_deg_accum<<<cdiv(nEdges, 256), 256, 0, stream>>>(dsts, dinv, nEdges);
    k_rsqrt_inplace<<<cdiv(nNodes, 256), 256, 0, stream>>>(dinv, nNodes);

    // --- input features -> f16 ---
    k_f32_to_f16<false><<<cdiv(totalElems, 256), 256, 0, stream>>>(x, hf16, totalElems);

    // --- layer 1 ---
    k_pack_w<<<DIM * DIM / 256, 256, 0, stream>>>(W1, Bpk);
    k_gemm_strip<<<gemmBlocks, 256, 0, stream>>>(hf16, Bpk, dinv, b1, tbuf, hA, nStrips);
    k_edge_scatter<<<cdiv(nEdges * 32, 256), 256, 0, stream>>>(tbuf, srcs, dsts, dinv, hA, nEdges);
    k_f32_to_f16<true><<<cdiv(totalElems, 256), 256, 0, stream>>>(hA, hf16, totalElems);

    // --- layer 2 ---
    k_pack_w<<<DIM * DIM / 256, 256, 0, stream>>>(W2, Bpk);
    k_gemm_strip<<<gemmBlocks, 256, 0, stream>>>(hf16, Bpk, dinv, b2, tbuf, hA, nStrips);
    k_edge_scatter<<<cdiv(nEdges * 32, 256), 256, 0, stream>>>(tbuf, srcs, dsts, dinv, hA, nEdges);
    k_f32_to_f16<true><<<cdiv(totalElems, 256), 256, 0, stream>>>(hA, hf16, totalElems);

    // --- layer 3 (no relu; keep f32 for pooling) ---
    k_pack_w<<<DIM * DIM / 256, 256, 0, stream>>>(W3, Bpk);
    k_gemm_strip<<<gemmBlocks, 256, 0, stream>>>(hf16, Bpk, dinv, b3, tbuf, hB, nStrips);
    k_edge_scatter<<<cdiv(nEdges * 32, 256), 256, 0, stream>>>(tbuf, srcs, dsts, dinv, hB, nEdges);

    // --- mean pool + projection ---
    k_zero<<<cdiv(nGraph * DIM, 256), 256, 0, stream>>>(pool, nGraph * DIM);
    k_zero<<<1, 256, 0, stream>>>(cnt, nGraph);
    k_pool_accum<<<cdiv(nNodes * 32, 256), 256, 0, stream>>>(hB, batch, pool, nNodes);
    k_cnt_accum<<<cdiv(nNodes, 256), 256, 0, stream>>>(batch, cnt, nNodes);
    k_final<<<nGraph, DIM, 0, stream>>>(pool, cnt, Wp, bp, out);
}
